// MultiHeadAttention_32847909879921
// MI455X (gfx1250) — compile-verified, matrix operand order checked by reading, not run on hardware
//
#include <hip/hip_runtime.h>

// ---------------------------------------------------------------------------
// MHA for MI455X (gfx1250): bf16 WMMA GEMMs + flash attention, f32 accumulate.
// B=2, S=2048, D=1024, H=16, HD=64.
// Data movement: double-buffered GLOBAL_LOAD_ASYNC_TO_LDS_B128 (ASYNCcnt)
// overlapping the v_wmma_f32_16x16x32_bf16 pipeline.
// ---------------------------------------------------------------------------

typedef __attribute__((ext_vector_type(16))) __bf16 v16bf;
typedef __attribute__((ext_vector_type(8)))  float  v8f;

union ABFrag { v16bf v; uint4 u[2]; };

__device__ __forceinline__ v8f wmma_bf16(v16bf a, v16bf b, v8f c) {
  // D = A(16x32 bf16) * B(32x16 bf16) + C(16x16 f32)
  return __builtin_amdgcn_wmma_f32_16x16x32_bf16(
      /*neg_a=*/false, a, /*neg_b=*/false, b,
      /*c_mod=*/(short)0, c, /*reuse_a=*/false, /*reuse_b=*/false);
}

// LDS byte offset of a shared-memory pointer (generic -> addrspace(3)).
__device__ __forceinline__ unsigned lds_off(const void* p) {
  typedef const __attribute__((address_space(3))) char AS3char;
  return (unsigned)(unsigned long long)(AS3char*)p;
}

// Async copy 16B global -> LDS per lane (GVS mode: sgpr base + vgpr offset).
// Tracked by ASYNCcnt, not LOADcnt: frees the wave to keep issuing WMMAs.
__device__ __forceinline__ void async_b128(unsigned lds_byte, const void* base,
                                           unsigned voff_bytes) {
  asm volatile("global_load_async_to_lds_b128 %0, %1, %2"
               :
               : "v"(lds_byte), "v"(voff_bytes), "s"(base)
               : "memory");
}
__device__ __forceinline__ void wait_async0() {
  asm volatile("s_wait_asynccnt 0" ::: "memory");
}

// ---------------------------------------------------------------------------
// fp32 -> bf16 cast (vectorized: float4 in, 4x bf16 out)
// ---------------------------------------------------------------------------
__global__ void f32_to_bf16_kernel(const float* __restrict__ in,
                                   __bf16* __restrict__ out, long n) {
  long i = ((long)blockIdx.x * blockDim.x + threadIdx.x) * 4;
  const long stride = (long)gridDim.x * blockDim.x * 4;
  for (; i < n; i += stride) {
    const float4 f = *(const float4*)(in + i);
    union { __bf16 e[4]; uint2 u; } o;
    o.e[0] = (__bf16)f.x; o.e[1] = (__bf16)f.y;
    o.e[2] = (__bf16)f.z; o.e[3] = (__bf16)f.w;
    *(uint2*)(out + i) = o.u;
  }
}

// ---------------------------------------------------------------------------
// GEMM: Out[m,n] = (sum_k A[m,k]*W[n,k] + bias[n]) * scaleOut
//   M = 4096 (b,s), N = K = 1024.  Block = 256 threads = 8 waves.
//   Tile: BM=64, BN=64, BK=32; double-buffered async LDS staging.
//   A_HEAD: A is stored [B,H,S,HD] (head-interleaved), else [M,K] row-major.
//   OUT_HEAD: Out is bf16 [B,H,S,HD], else f32 [M,N] row-major.
// ---------------------------------------------------------------------------
template <int OUT_HEAD, int A_HEAD>
__global__ __launch_bounds__(256) void gemm_wmma(
    const __bf16* __restrict__ A, const __bf16* __restrict__ W,
    const float* __restrict__ bias, void* __restrict__ Out, float scaleOut) {
  constexpr int K = 1024, N = 1024;
  __shared__ __align__(16) __bf16 sA[2][64 * 32];
  __shared__ __align__(16) __bf16 sB[2][64 * 32];

  const int tid  = threadIdx.x;
  const int n0   = blockIdx.x * 64;
  const int m0   = blockIdx.y * 64;
  const int wid  = tid >> 5;
  const int lane = tid & 31;
  const int lr   = lane & 15;
  const int khalf = (lane >> 4) * 8;   // k-offset for this lane group
  const int wm   = (wid >> 1) * 16;    // wave row offset in tile
  const int wn   = (wid & 1) * 32;     // wave col offset in tile

  const int ldRow = tid >> 2;          // 0..63
  const int ldCol = (tid & 3) * 8;     // 0,8,16,24

  // Per-thread async staging of one 16B chunk of A and of B.
  auto issue_tile = [&](int buf, int kk) {
    const int ak = kk + ldCol;
    unsigned aByte;
    if (A_HEAD) {
      const int m = m0 + ldRow;
      const int b = m >> 11, s = m & 2047;
      const int h = ak >> 6, hd = ak & 63;
      aByte = (unsigned)((((long)((b << 4) | h) * 2048 + s) * 64 + hd) * 2);
    } else {
      aByte = (unsigned)(((long)(m0 + ldRow) * K + ak) * 2);
    }
    async_b128(lds_off(&sA[buf][ldRow * 32 + ldCol]), A, aByte);
    async_b128(lds_off(&sB[buf][ldRow * 32 + ldCol]), W,
               (unsigned)(((long)(n0 + ldRow) * K + ak) * 2));
  };

  v8f acc0 = {}, acc1 = {};

  issue_tile(0, 0);
  wait_async0();
  __syncthreads();

  int cur = 0;
  for (int kk = 0; kk < K; kk += 32) {
    if (kk + 32 < K) issue_tile(cur ^ 1, kk + 32);  // overlap next-tile copy

    ABFrag a, b0, b1;
    const int abase = (wm + lr) * 32 + khalf;
    a.u[0] = *(const uint4*)&sA[cur][abase];
    a.u[1] = *(const uint4*)&sA[cur][abase + 16];
    const int bbase0 = (wn + lr) * 32 + khalf;
    b0.u[0] = *(const uint4*)&sB[cur][bbase0];
    b0.u[1] = *(const uint4*)&sB[cur][bbase0 + 16];
    const int bbase1 = (wn + 16 + lr) * 32 + khalf;
    b1.u[0] = *(const uint4*)&sB[cur][bbase1];
    b1.u[1] = *(const uint4*)&sB[cur][bbase1 + 16];

    acc0 = wmma_bf16(a.v, b0.v, acc0);
    acc1 = wmma_bf16(a.v, b1.v, acc1);

    wait_async0();      // next tile landed
    __syncthreads();    // single barrier per K-step
    cur ^= 1;
  }

  // ---- epilogue: bias, scale, store ----
  const int rowHalf = khalf;  // (lane>>4)*8 row offset in C layout
#pragma unroll
  for (int f = 0; f < 2; ++f) {
    v8f acc = f ? acc1 : acc0;
    const int col = n0 + wn + f * 16 + lr;
    const float bv = bias[col];
#pragma unroll
    for (int r = 0; r < 8; ++r) {
      const int m = m0 + wm + rowHalf + r;
      const float val = (acc[r] + bv) * scaleOut;
      if (OUT_HEAD) {
        const int b = m >> 11, s = m & 2047;
        const int h = col >> 6, hd = col & 63;
        ((__bf16*)Out)[((long)((b << 4) | h) * 2048 + s) * 64 + hd] =
            (__bf16)val;
      } else {
        ((float*)Out)[(long)m * N + col] = val;
      }
    }
  }
}

// ---------------------------------------------------------------------------
// Flash attention with causal mask, per (b, h, 64-query tile).
// Block = 128 threads = 4 waves; wave w handles query rows [qt*64+16w, +16).
// K tiles: double-buffered async global->LDS.  V tiles: global loads issued
// before the compute phase, transposed into LDS after it (overlap).
// Q pre-scaled by 1/sqrt(HD) in the Q GEMM.
// ---------------------------------------------------------------------------
__global__ __launch_bounds__(128) void flash_attn_kernel(
    const __bf16* __restrict__ Q, const __bf16* __restrict__ Kt,
    const __bf16* __restrict__ V, __bf16* __restrict__ O) {
  constexpr int S = 2048, HD = 64, H = 16;
  __shared__ __align__(16) __bf16 sK[2][64 * 64];   // [key][hd]
  __shared__ __align__(16) __bf16 sVt[2][64 * 64];  // [hd][key] (transposed)
  __shared__ __align__(16) __bf16 sP[4 * 16 * 64];  // wave-private P staging

  const int qt   = blockIdx.x;
  const int h    = blockIdx.y;
  const int b    = blockIdx.z;
  const int tid  = threadIdx.x;
  const int wid  = tid >> 5;
  const int lane = tid & 31;
  const int lr   = lane & 15;
  const int khalf = (lane >> 4) * 8;

  const long headBase = ((long)(b * H + h)) * S * HD;
  const __bf16* Kh = Kt + headBase;
  const __bf16* Vh = V + headBase;
  const int qrow0 = qt * 64 + wid * 16;

  // Chunk decomposition for tile staging: 512 16B chunks, 4 per thread.
  int cRow[4], cCol[4];
#pragma unroll
  for (int p = 0; p < 4; ++p) {
    const int chunk = tid + p * 128;
    cRow[p] = chunk >> 3;          // key 0..63
    cCol[p] = (chunk & 7) * 8;     // hd chunk
  }

  // Q fragments, held in registers for the whole kernel (2 k-steps over HD).
  ABFrag qf[2];
#pragma unroll
  for (int t = 0; t < 2; ++t) {
    const __bf16* qp = Q + headBase + (long)(qrow0 + lr) * HD + t * 32 + khalf;
    qf[t].u[0] = *(const uint4*)qp;
    qf[t].u[1] = *(const uint4*)(qp + 16);
  }

  v8f oacc[4] = {{}, {}, {}, {}};
  float mrow[8], lrow[8];
#pragma unroll
  for (int r = 0; r < 8; ++r) { mrow[r] = -1e30f; lrow[r] = 0.0f; }

  const int ntiles = qt + 1;  // causal: only key tiles <= query tile

  // ---- prologue: stage tile 0 ----
  {
#pragma unroll
    for (int p = 0; p < 4; ++p)
      async_b128(lds_off(&sK[0][cRow[p] * 64 + cCol[p]]), Kh,
                 (unsigned)(((long)cRow[p] * HD + cCol[p]) * 2));
#pragma unroll
    for (int p = 0; p < 4; ++p) {
      union { uint4 u; __bf16 e[8]; } vv;
      vv.u = *(const uint4*)&Vh[(long)cRow[p] * HD + cCol[p]];
#pragma unroll
      for (int e = 0; e < 8; ++e)
        sVt[0][(cCol[p] + e) * 64 + cRow[p]] = vv.e[e];
    }
    wait_async0();
    __syncthreads();
  }

  for (int ktile = 0; ktile < ntiles; ++ktile) {
    const int k0 = ktile * 64;
    const int cur = ktile & 1, nxt = cur ^ 1;
    const bool havenext = (ktile + 1) < ntiles;

    // ---- issue next tile's copies before computing (overlap) ----
    uint4 vbuf[4];
    if (havenext) {
      const int k1 = k0 + 64;
#pragma unroll
      for (int p = 0; p < 4; ++p)
        async_b128(lds_off(&sK[nxt][cRow[p] * 64 + cCol[p]]), Kh,
                   (unsigned)(((long)(k1 + cRow[p]) * HD + cCol[p]) * 2));
#pragma unroll
      for (int p = 0; p < 4; ++p)
        vbuf[p] = *(const uint4*)&Vh[(long)(k1 + cRow[p]) * HD + cCol[p]];
    }

    // ---- scores: S(16x64) = Q(16x64) @ K^T, 2 k-steps per 16 keys ----
    v8f sc[4];
#pragma unroll
    for (int f = 0; f < 4; ++f) {
      const int key = f * 16 + lr;
      ABFrag bk0, bk1;
      bk0.u[0] = *(const uint4*)&sK[cur][key * 64 + khalf];
      bk0.u[1] = *(const uint4*)&sK[cur][key * 64 + khalf + 16];
      bk1.u[0] = *(const uint4*)&sK[cur][key * 64 + 32 + khalf];
      bk1.u[1] = *(const uint4*)&sK[cur][key * 64 + 32 + khalf + 16];
      v8f c = {};
      c = wmma_bf16(qf[0].v, bk0.v, c);
      c = wmma_bf16(qf[1].v, bk1.v, c);
      sc[f] = c;
    }

    // ---- causal mask + online softmax (row reductions across 16 lanes) ----
    float mnew[8];
#pragma unroll
    for (int r = 0; r < 8; ++r) {
      const int qglob = qrow0 + khalf + r;
      float vmax = -1e30f;
#pragma unroll
      for (int f = 0; f < 4; ++f) {
        const int key = k0 + f * 16 + lr;
        float s = sc[f][r];
        if (key > qglob) s = -1e9f;
        sc[f][r] = s;
        vmax = fmaxf(vmax, s);
      }
#pragma unroll
      for (int m = 1; m < 16; m <<= 1)
        vmax = fmaxf(vmax, __shfl_xor(vmax, m, 32));
      mnew[r] = fmaxf(mrow[r], vmax);
    }
#pragma unroll
    for (int r = 0; r < 8; ++r) {
      const float alpha = __expf(mrow[r] - mnew[r]);
      float rsum = 0.0f;
#pragma unroll
      for (int f = 0; f < 4; ++f) {
        const float p = __expf(sc[f][r] - mnew[r]);
        sc[f][r] = p;
        rsum += p;
      }
#pragma unroll
      for (int m = 1; m < 16; m <<= 1) rsum += __shfl_xor(rsum, m, 32);
      lrow[r] = lrow[r] * alpha + rsum;
      mrow[r] = mnew[r];
#pragma unroll
      for (int f = 0; f < 4; ++f) oacc[f][r] *= alpha;
    }

    // ---- restage P through wave-private LDS (C-layout -> A-layout) ----
    __bf16* pw = &sP[wid * 1024];
#pragma unroll
    for (int f = 0; f < 4; ++f)
#pragma unroll
      for (int r = 0; r < 8; ++r)
        pw[(khalf + r) * 64 + f * 16 + lr] = (__bf16)sc[f][r];
    asm volatile("s_wait_dscnt 0" ::: "memory");  // in-wave LDS RAW

    ABFrag pa[2];
#pragma unroll
    for (int t = 0; t < 2; ++t) {
      const __bf16* pp = pw + lr * 64 + t * 32 + khalf;
      pa[t].u[0] = *(const uint4*)pp;
      pa[t].u[1] = *(const uint4*)(pp + 16);
    }

    // ---- O(16x64) += P(16x64) @ V(64x64), 2 k-steps per 16 hd cols ----
#pragma unroll
    for (int f2 = 0; f2 < 4; ++f2) {
      const int hd = f2 * 16 + lr;
      ABFrag bv0, bv1;
      bv0.u[0] = *(const uint4*)&sVt[cur][hd * 64 + khalf];
      bv0.u[1] = *(const uint4*)&sVt[cur][hd * 64 + khalf + 16];
      bv1.u[0] = *(const uint4*)&sVt[cur][hd * 64 + 32 + khalf];
      bv1.u[1] = *(const uint4*)&sVt[cur][hd * 64 + 32 + khalf + 16];
      oacc[f2] = wmma_bf16(pa[0].v, bv0.v, oacc[f2]);
      oacc[f2] = wmma_bf16(pa[1].v, bv1.v, oacc[f2]);
    }

    // ---- scatter next V tile (transposed) into the other buffer ----
    if (havenext) {
#pragma unroll
      for (int p = 0; p < 4; ++p) {
        union { uint4 u; __bf16 e[8]; } vv;
        vv.u = vbuf[p];
#pragma unroll
        for (int e = 0; e < 8; ++e)
          sVt[nxt][(cCol[p] + e) * 64 + cRow[p]] = vv.e[e];
      }
    }
    wait_async0();
    __syncthreads();
  }

  // ---- normalize and store O [B,H,S,HD] bf16 ----
#pragma unroll
  for (int r = 0; r < 8; ++r) {
    const float rinv = 1.0f / lrow[r];
    const int q = qrow0 + khalf + r;
#pragma unroll
    for (int f2 = 0; f2 < 4; ++f2) {
      const int hd = f2 * 16 + lr;
      O[headBase + (long)q * HD + hd] = (__bf16)(oacc[f2][r] * rinv);
    }
  }
}

// ---------------------------------------------------------------------------
// Launch
// ---------------------------------------------------------------------------
extern "C" void kernel_launch(void* const* d_in, const int* in_sizes, int n_in,
                              void* d_out, int out_size, void* d_ws,
                              size_t ws_size, hipStream_t stream) {
  const float* x  = (const float*)d_in[0];
  const float* wq = (const float*)d_in[1];
  const float* bq = (const float*)d_in[2];
  const float* wk = (const float*)d_in[3];
  const float* bk = (const float*)d_in[4];
  const float* wv = (const float*)d_in[5];
  const float* bv = (const float*)d_in[6];
  const float* wo = (const float*)d_in[7];
  const float* bo = (const float*)d_in[8];

  const long XB = 4194304;  // B*S*D
  const long WN = 1048576;  // D*D

  __bf16* ws  = (__bf16*)d_ws;
  __bf16* xb  = ws;
  __bf16* wqb = xb + XB;
  __bf16* wkb = wqb + WN;
  __bf16* wvb = wkb + WN;
  __bf16* wob = wvb + WN;
  __bf16* Qb  = wob + WN;
  __bf16* Kb  = Qb + XB;
  __bf16* Vb  = Kb + XB;
  __bf16* Ob  = Vb + XB;

  f32_to_bf16_kernel<<<1024, 256, 0, stream>>>(x, xb, XB);
  f32_to_bf16_kernel<<<256, 256, 0, stream>>>(wq, wqb, WN);
  f32_to_bf16_kernel<<<256, 256, 0, stream>>>(wk, wkb, WN);
  f32_to_bf16_kernel<<<256, 256, 0, stream>>>(wv, wvb, WN);
  f32_to_bf16_kernel<<<256, 256, 0, stream>>>(wo, wob, WN);

  const dim3 gg(16, 64), gb(256);
  // Q gets the 1/sqrt(HD)=0.125 attention scale folded in.
  gemm_wmma<1, 0><<<gg, gb, 0, stream>>>(xb, wqb, bq, Qb, 0.125f);
  gemm_wmma<1, 0><<<gg, gb, 0, stream>>>(xb, wkb, bk, Kb, 1.0f);
  gemm_wmma<1, 0><<<gg, gb, 0, stream>>>(xb, wvb, bv, Vb, 1.0f);

  flash_attn_kernel<<<dim3(32, 16, 2), 128, 0, stream>>>(Qb, Kb, Vb, Ob);

  gemm_wmma<0, 1><<<gg, gb, 0, stream>>>(Ob, wob, bo, d_out, 1.0f);
}
